// GCN_DECONF_33827162423508
// MI455X (gfx1250) — matrix operand order repaired and checked by reading, not compiled, not run on hardware
//
#include <hip/hip_runtime.h>
#include <hip/hip_bf16.h>
#include <math.h>
#include <stdint.h>

#define NNODES 10000
#define NFEAT  128
#define NHID   64
#define MAXNNZ 128

typedef __attribute__((ext_vector_type(16))) _Float16 v16h;
typedef __attribute__((ext_vector_type(8)))  float    v8f;

// --- CDNA5 async global->LDS (ASYNCcnt) path, guarded ----------------------
#if defined(__has_builtin)
#  if __has_builtin(__builtin_amdgcn_global_load_async_to_lds_b32) && \
      __has_builtin(__builtin_amdgcn_s_wait_asynccnt)
#    define HAVE_ASYNC_LDS 1
#  endif
#endif

__device__ __forceinline__ void async_copy_b32(void* lds_dst, const void* gsrc) {
#ifdef HAVE_ASYNC_LDS
    __builtin_amdgcn_global_load_async_to_lds_b32(
        (__attribute__((address_space(1))) int*)(uintptr_t)gsrc,
        (__attribute__((address_space(3))) int*)(uintptr_t)lds_dst, 0, 0);
#else
    *(uint32_t*)lds_dst = *(const uint32_t*)gsrc;
#endif
}
__device__ __forceinline__ void async_wait_all() {
#ifdef HAVE_ASYNC_LDS
    __builtin_amdgcn_s_wait_asynccnt(0);
#endif
}

// ---------------------------------------------------------------------------
// Kernel 1: stream adj once (400MB -> HBM-bound floor ~17us), extract CSR.
// Deterministic, column-ordered compaction via wave-scan + block-scan.
// ---------------------------------------------------------------------------
__global__ void build_csr(const float* __restrict__ adj,
                          float* __restrict__ val, int* __restrict__ col,
                          int* __restrict__ cnt) {
    const int i    = blockIdx.x;
    const int tid  = threadIdx.x;          // 0..255
    const int lane = tid & 31;
    const int wv   = tid >> 5;             // 8 waves (wave32)
    __shared__ int wsum[8];
    __shared__ int base;
    if (tid == 0) base = 0;
    __syncthreads();

    const float4* row = (const float4*)(adj + (size_t)i * NNODES);
    for (int q0 = 0; q0 < 2500; q0 += 256) {
        int  q      = q0 + tid;
        bool active = q < 2500;
        float4 v = active ? row[q] : make_float4(0.f, 0.f, 0.f, 0.f);
        if (active) __builtin_prefetch(row + q + 256, 0, 0);  // global_prefetch_b8

        int f0 = (active && v.x != 0.f), f1 = (active && v.y != 0.f);
        int f2 = (active && v.z != 0.f), f3 = (active && v.w != 0.f);
        int c  = f0 + f1 + f2 + f3;
        // inclusive scan within wave32
        int sc = c;
        #pragma unroll
        for (int off = 1; off < 32; off <<= 1) {
            int n = __shfl_up(sc, off);
            if (lane >= off) sc += n;
        }
        if (lane == 31) wsum[wv] = sc;
        __syncthreads();
        int wbase = 0, total = 0;
        #pragma unroll
        for (int w = 0; w < 8; ++w) { if (w < wv) wbase += wsum[w]; total += wsum[w]; }
        int pos = base + wbase + (sc - c);   // exclusive prefix for this thread
        int j   = 4 * q;
        if (f0) { if (pos < MAXNNZ) { col[(size_t)i*MAXNNZ+pos] = j;   val[(size_t)i*MAXNNZ+pos] = v.x; } pos++; }
        if (f1) { if (pos < MAXNNZ) { col[(size_t)i*MAXNNZ+pos] = j+1; val[(size_t)i*MAXNNZ+pos] = v.y; } pos++; }
        if (f2) { if (pos < MAXNNZ) { col[(size_t)i*MAXNNZ+pos] = j+2; val[(size_t)i*MAXNNZ+pos] = v.z; } pos++; }
        if (f3) { if (pos < MAXNNZ) { col[(size_t)i*MAXNNZ+pos] = j+3; val[(size_t)i*MAXNNZ+pos] = v.w; } pos++; }
        __syncthreads();
        if (tid == 0) base += total;
    }
    __syncthreads();
    if (tid == 0) cnt[i] = base > MAXNNZ ? MAXNNZ : base;
}

// ---------------------------------------------------------------------------
// WMMA GEMM: C[M,64] = A[M,K] @ W[K,64] (+bias)(+relu). One wave per 16x16
// tile, fully unrolled K -> K/32 back-to-back v_wmma_f32_16x16x32_f16 with
// fp32 accumulate. M%16==0, K in {64,128}.
// EPI: 0 = none, 1 = +bias, 2 = relu(+bias)
// ---------------------------------------------------------------------------
template <int K, int EPI>
__global__ void gemm_wmma(const float* __restrict__ A, int lda,
                          const float* __restrict__ W, int ldw,
                          const float* __restrict__ bias,
                          float* __restrict__ C, int ldc) {
    const int lane = threadIdx.x;        // blockDim = 32 (one wave)
    const int m0   = blockIdx.x * 16;
    const int n0   = blockIdx.y * 16;
    const int row  = lane & 15;
    const int hi   = lane >> 4;          // lane group 0 / 1

    v8f acc = {};
    #pragma unroll
    for (int k0 = 0; k0 < K; k0 += 32) {
        v16h a, b;
        // A fragment (16x32 f16): lane holds M=row; K per ISA 16-bit A layout
        #pragma unroll
        for (int v = 0; v < 8; ++v) {
            int kb = ((v < 4) ? 2 * v : 16 + 2 * (v - 4)) + (hi ? 8 : 0);
            float2 f2 = *(const float2*)(A + (size_t)(m0 + row) * lda + k0 + kb);
            a[2 * v]     = (_Float16)f2.x;
            a[2 * v + 1] = (_Float16)f2.y;
        }
        // B fragment (32x16 f16): lane holds N=row; K per ISA 16-bit layout (same as A)
        #pragma unroll
        for (int v = 0; v < 8; ++v) {
            int kk = ((v < 4) ? 2 * v : 16 + 2 * (v - 4)) + (hi ? 8 : 0);
            b[2 * v]     = (_Float16)W[(size_t)(k0 + kk)     * ldw + n0 + row];
            b[2 * v + 1] = (_Float16)W[(size_t)(k0 + kk + 1) * ldw + n0 + row];
        }
        acc = __builtin_amdgcn_wmma_f32_16x16x32_f16(false, a, false, b,
                                                     (short)0, acc, false, false);
    }
    float bv = (EPI != 0) ? bias[n0 + row] : 0.f;
    #pragma unroll
    for (int r = 0; r < 8; ++r) {
        int   m = m0 + r + hi * 8;
        float o = acc[r] + bv;
        if (EPI == 2) o = fmaxf(o, 0.f);
        C[(size_t)m * ldc + n0 + row] = o;
    }
}

// ---------------------------------------------------------------------------
// SpMM + bias + relu: R[i,0:128] = relu( sum_k val * Z[col,:] + [bg|bt] )
// One block of 128 threads per row; CSR row staged in LDS via async copies.
// ---------------------------------------------------------------------------
__global__ void spmm_bias_relu(const float* __restrict__ Z,
                               const float* __restrict__ val,
                               const int* __restrict__ col,
                               const int* __restrict__ cnt,
                               const float* __restrict__ bg,
                               const float* __restrict__ bt,
                               float* __restrict__ R) {
    __shared__ float sv[MAXNNZ];
    __shared__ int   sc[MAXNNZ];
    const int i   = blockIdx.x;
    const int tid = threadIdx.x;           // 0..127 = output column
    const int n   = cnt[i];
    if (tid < n) {
        async_copy_b32(&sv[tid], &val[(size_t)i * MAXNNZ + tid]);
        async_copy_b32(&sc[tid], &col[(size_t)i * MAXNNZ + tid]);
    }
    async_wait_all();
    __syncthreads();
    float acc = 0.f;
    for (int k = 0; k < n; ++k) acc += sv[k] * Z[(size_t)sc[k] * 128 + tid];
    float b = (tid < 64) ? bg[tid] : bt[tid - 64];
    R[(size_t)i * 128 + tid] = fmaxf(acc + b, 0.f);
}

// ---------------------------------------------------------------------------
// s_src[i] = R[i,:]·a[0:128], s_dst[i] = R[i,:]·a[128:256]; wave per row.
// ---------------------------------------------------------------------------
__global__ void scores_kernel(const float* __restrict__ R, const float* __restrict__ a,
                              float* __restrict__ ssrc, float* __restrict__ sdst) {
    const int i    = blockIdx.x * 8 + (threadIdx.x >> 5);
    const int lane = threadIdx.x & 31;
    float4 r  = *(const float4*)(R + (size_t)i * 128 + lane * 4);
    float4 as = *(const float4*)(a + lane * 4);
    float4 ad = *(const float4*)(a + 128 + lane * 4);
    float s0 = r.x*as.x + r.y*as.y + r.z*as.z + r.w*as.w;
    float s1 = r.x*ad.x + r.y*ad.y + r.z*ad.z + r.w*ad.w;
    #pragma unroll
    for (int off = 16; off > 0; off >>= 1) {
        s0 += __shfl_xor(s0, off);
        s1 += __shfl_xor(s1, off);
    }
    if (lane == 0) { ssrc[i] = s0; sdst[i] = s1; }
}

// colsum(rep_t): deterministic two-stage reduction
__global__ void colsum_part(const float* __restrict__ R, float* __restrict__ part) {
    const int b = blockIdx.x;   // 0..63
    const int h = threadIdx.x;  // 0..63
    float acc = 0.f;
    for (int r = b; r < NNODES; r += 64) acc += R[(size_t)r * 128 + 64 + h];
    part[b * 64 + h] = acc;
}
__global__ void colsum_final(const float* __restrict__ part, float* __restrict__ csum) {
    const int h = threadIdx.x;  // 0..63
    float acc = 0.f;
    for (int b = 0; b < 64; ++b) acc += part[b * 64 + h];
    csum[h] = acc;
}

// ---------------------------------------------------------------------------
// Attention row softmax (dense-row semantics collapsed analytically) +
// rep_out = att @ rep_t + rep_o. One block of 128 threads per row.
// ---------------------------------------------------------------------------
__global__ void attn_kernel(const float* __restrict__ R,
                            const int* __restrict__ col, const int* __restrict__ cnt,
                            const float* __restrict__ ssrc, const float* __restrict__ sdst,
                            const float* __restrict__ csum,
                            float* __restrict__ rep_out) {
    __shared__ float ev[MAXNNZ];
    __shared__ int   cl[MAXNNZ];
    __shared__ float red[128];
    const int i   = blockIdx.x;
    const int tid = threadIdx.x;
    const int n   = cnt[i];
    const float si = ssrc[i];
    if (tid < n) async_copy_b32(&cl[tid], &col[(size_t)i * MAXNNZ + tid]);
    async_wait_all();
    __syncthreads();
    float s = 0.f;
    if (tid < n) { s = si + sdst[cl[tid]]; ev[tid] = s; }
    red[tid] = (tid < n) ? fmaxf(s, 0.f) : 0.f;   // implicit zeros dominate start
    __syncthreads();
    for (int st = 64; st > 0; st >>= 1) {
        if (tid < st) red[tid] = fmaxf(red[tid], red[tid + st]);
        __syncthreads();
    }
    const float m = red[0];
    __syncthreads();
    float e = (tid < n) ? expf(ev[tid] - m) : 0.f;
    if (tid < n) ev[tid] = e;
    red[tid] = e;
    __syncthreads();
    for (int st = 64; st > 0; st >>= 1) {
        if (tid < st) red[tid] += red[tid + st];
        __syncthreads();
    }
    const float zt = expf(-m);
    const float Zd = red[0] + (float)(NNODES - n) * zt;   // full dense-row denom
    if (tid < 64) {
        float num = 0.f, ts = 0.f;
        for (int k = 0; k < n; ++k) {
            float rt = R[(size_t)cl[k] * 128 + 64 + tid];
            num += ev[k] * rt;
            ts  += rt;
        }
        num += zt * (csum[tid] - ts);                      // implicit-zero columns
        rep_out[(size_t)i * 64 + tid] = num / Zd + R[(size_t)i * 128 + tid];
    }
}

// treatment = sigmoid(hidT @ Wpp2 + bpp2), Wpp2:[64,2]
__global__ void treat_kernel(const float* __restrict__ hid,
                             const float* __restrict__ Wpp2, const float* __restrict__ bpp2,
                             float* __restrict__ out) {
    int i = blockIdx.x * blockDim.x + threadIdx.x;
    if (i >= NNODES) return;
    float a0 = bpp2[0], a1 = bpp2[1];
    for (int h = 0; h < 64; ++h) {
        float v = hid[(size_t)i * 64 + h];
        a0 += v * Wpp2[h * 2 + 0];
        a1 += v * Wpp2[h * 2 + 1];
    }
    out[(size_t)i * 2 + 0] = 1.f / (1.f + expf(-a0));
    out[(size_t)i * 2 + 1] = 1.f / (1.f + expf(-a1));
}

// y[i] = t>0 ? yb1@Wo1+bo1 : yb0@Wo0+bo0
__global__ void yout_kernel(const float* __restrict__ yb0, const float* __restrict__ yb1,
                            const float* __restrict__ Wo0, const float* __restrict__ bo0,
                            const float* __restrict__ Wo1, const float* __restrict__ bo1,
                            const int* __restrict__ t, float* __restrict__ y) {
    int i = blockIdx.x * blockDim.x + threadIdx.x;
    if (i >= NNODES) return;
    float a0 = bo0[0], a1 = bo1[0];
    for (int h = 0; h < 64; ++h) {
        a0 += yb0[(size_t)i * 64 + h] * Wo0[h];
        a1 += yb1[(size_t)i * 64 + h] * Wo1[h];
    }
    y[i] = (t[i] > 0) ? a1 : a0;
}

// ---------------------------------------------------------------------------
extern "C" void kernel_launch(void* const* d_in, const int* in_sizes, int n_in,
                              void* d_out, int out_size, void* d_ws, size_t ws_size,
                              hipStream_t stream) {
    (void)in_sizes; (void)n_in; (void)out_size; (void)ws_size;
    const float* x    = (const float*)d_in[0];
    const float* adj  = (const float*)d_in[1];
    const int*   t    = (const int*)  d_in[2];
    const float* Wg0  = (const float*)d_in[3],  *bg0  = (const float*)d_in[4];
    const float* Wg1  = (const float*)d_in[5],  *bg1  = (const float*)d_in[6];
    const float* Wt0  = (const float*)d_in[7],  *bt0  = (const float*)d_in[8];
    const float* Wt1  = (const float*)d_in[9],  *bt1  = (const float*)d_in[10];
    const float* W000 = (const float*)d_in[11], *b000 = (const float*)d_in[12];
    const float* W001 = (const float*)d_in[13], *b001 = (const float*)d_in[14];
    const float* W100 = (const float*)d_in[15], *b100 = (const float*)d_in[16];
    const float* W101 = (const float*)d_in[17], *b101 = (const float*)d_in[18];
    const float* Wo0  = (const float*)d_in[19], *bo0  = (const float*)d_in[20];
    const float* Wo1  = (const float*)d_in[21], *bo1  = (const float*)d_in[22];
    const float* Wpp  = (const float*)d_in[23], *bpp  = (const float*)d_in[24];
    const float* Wpp2 = (const float*)d_in[25], *bpp2 = (const float*)d_in[26];
    const float* a    = (const float*)d_in[27];

    float* ws    = (float*)d_ws;
    float* Z     = ws;                       // [10000,128]
    float* R     = ws + 1280000;             // [10000,128]  ([rep_o | rep_t])
    float* VAL   = ws + 2560000;             // [10000,128]
    int*   COL   = (int*)(ws + 3840000);     // [10000,128]
    int*   CNT   = (int*)(ws + 5120000);     // [10000]
    float* SSRC  = ws + 5130000;             // [10000]
    float* SDST  = ws + 5140000;             // [10000]
    float* CSUM  = ws + 5150000;             // [64]
    float* CPART = ws + 5150064;             // [64*64]
    float* HIDT  = ws + 5154160;             // [10000,64]
    float* YB0   = ws + 5794160;             // [10000,64]
    float* YB1   = ws + 6434160;             // [10000,64]
    float* YB2   = VAL;                      // CSR vals dead after attn -> reuse
    float* YB3   = VAL + 640000;

    float* y_out   = (float*)d_out;          // [10000]
    float* rep_out = y_out + 10000;          // [10000,64]
    float* treat   = y_out + 650000;         // [10000,2]

    const dim3 gemm_grid(NNODES / 16, 64 / 16);

    // 1) single HBM pass over adj -> CSR (bandwidth floor of the whole net)
    build_csr<<<NNODES, 256, 0, stream>>>(adj, VAL, COL, CNT);

    // 2) layer-1 feature GEMMs: Z = [x@Wg0 | x@Wt0]
    gemm_wmma<128, 0><<<gemm_grid, 32, 0, stream>>>(x, 128, Wg0, 64, nullptr, Z,      128);
    gemm_wmma<128, 0><<<gemm_grid, 32, 0, stream>>>(x, 128, Wt0, 64, nullptr, Z + 64, 128);
    // 3) R1 = relu(adj@Z + [bg0|bt0])
    spmm_bias_relu<<<NNODES, 128, 0, stream>>>(Z, VAL, COL, CNT, bg0, bt0, R);
    // 4) layer-2: Z = [R1_o@Wg1 | R1_t@Wt1]
    gemm_wmma<64, 0><<<gemm_grid, 32, 0, stream>>>(R,      128, Wg1, 64, nullptr, Z,      128);
    gemm_wmma<64, 0><<<gemm_grid, 32, 0, stream>>>(R + 64, 128, Wt1, 64, nullptr, Z + 64, 128);
    // 5) R = [rep_o | rep_t]
    spmm_bias_relu<<<NNODES, 128, 0, stream>>>(Z, VAL, COL, CNT, bg1, bt1, R);

    // 6) attention scalars and colsum(rep_t)
    scores_kernel<<<NNODES / 8, 256, 0, stream>>>(R, a, SSRC, SDST);
    colsum_part  <<<64, 64, 0, stream>>>(R, CPART);
    colsum_final <<<1, 64, 0, stream>>>(CPART, CSUM);
    // 7) rep_out = softmax_row(masked scores) @ rep_t + rep_o
    attn_kernel<<<NNODES, 128, 0, stream>>>(R, COL, CNT, SSRC, SDST, CSUM, rep_out);

    // 8) treatment head: sigmoid((rep_t@Wpp + bpp)@Wpp2 + bpp2)
    gemm_wmma<64, 1><<<gemm_grid, 32, 0, stream>>>(R + 64, 128, Wpp, 64, bpp, HIDT, 64);
    treat_kernel<<<(NNODES + 255) / 256, 256, 0, stream>>>(HIDT, Wpp2, bpp2, treat);

    // 9) outcome heads
    gemm_wmma<64, 2><<<gemm_grid, 32, 0, stream>>>(rep_out, 64, W000, 64, b000, YB0, 64);
    gemm_wmma<64, 2><<<gemm_grid, 32, 0, stream>>>(rep_out, 64, W100, 64, b100, YB1, 64);
    gemm_wmma<64, 2><<<gemm_grid, 32, 0, stream>>>(YB0,     64, W001, 64, b001, YB2, 64);
    gemm_wmma<64, 2><<<gemm_grid, 32, 0, stream>>>(YB1,     64, W101, 64, b101, YB3, 64);
    yout_kernel<<<(NNODES + 255) / 256, 256, 0, stream>>>(YB2, YB3, Wo0, bo0, Wo1, bo1, t, y_out);
}
